// MS_MSA_14010183319996
// MI455X (gfx1250) — compile-verified, matrix operand order checked by reading, not run on hardware
//
#include <hip/hip_runtime.h>
#include <math.h>

// MS-MSA fused pipeline for MI455X (gfx1250, wave32, WMMA bf16).
// Memory-bound (~37 GFLOP vs ~300MB HBM traffic @ 23.3 TB/s) -> bf16 operands,
// f32 WMMA accumulation, attention-apply+projection fused into per-batch W2.
//
// Workspace layout (bytes), total ~160.5 MB:
//   q16  [65536*256] bf16 : 0
//   k16  [65536*256] bf16 : 33554432
//   v16  [65536*256] bf16 : 67108864
//   t1   [65536*256] f32  : 100663296   (conv1+gelu intermediate)
//   ssq  [2*4*256]   f32  : 167772160   (sumsq_q | sumsq_k)
//   W2   [4*256*256] bf16 : 167780352

#define NTOK   16384          // 128*128
#define BATCH  4
#define HEADS  8
#define ROWS   (BATCH * NTOK) // 65536

typedef __attribute__((ext_vector_type(16))) __bf16         v16bf;
typedef __attribute__((ext_vector_type(8)))  float          v8f;
typedef __attribute__((ext_vector_type(8)))  unsigned short u16x8;

union Frag16 {
  v16bf v;
  u16x8 h[2];
  unsigned short u[16];
};

// Native conversions: let clang use hardware bf16 cvt where available.
__device__ __forceinline__ unsigned int pack2bf(float a, float b) {
  union { __bf16 b2[2]; unsigned int u; } cv;
  cv.b2[0] = (__bf16)a;
  cv.b2[1] = (__bf16)b;
  return cv.u;
}
__device__ __forceinline__ unsigned short f2bf(float f) {
  union { __bf16 b; unsigned short u; } cv;
  cv.b = (__bf16)f;
  return cv.u;
}
__device__ __forceinline__ float bf2f(unsigned short u) {
  return __uint_as_float(((unsigned int)u) << 16);
}

__global__ void k_zero(float* __restrict__ p, int n) {
  for (int i = threadIdx.x; i < n; i += 256) p[i] = 0.0f;
}

// ---------------------------------------------------------------------------
// GEMM: O16[r][n] = bf16( A[r][:] @ W[:][n] ), A f32 [ROWS][256], W f32 [256][256].
// Block tile 128(M)x128(N), 8 waves (4x2), wave tile 32x64, K chunks of 32.
// LDS tiles written as packed b32 (two bf16). Optional epilogue: token-axis
// sumsq[b*256+n] += sum_r O^2 (for the L2 norms over tokens).
// ---------------------------------------------------------------------------
__global__ __launch_bounds__(256) void k_gemm_qkv(
    const float* __restrict__ A, const float* __restrict__ W,
    unsigned short* __restrict__ O, float* __restrict__ sumsq) {
  __shared__ __align__(16) unsigned short lA[128 * 32];
  __shared__ __align__(16) unsigned short lB[128 * 32]; // transposed: [n][k]
  __shared__ float lSum[128];

  const int tid  = threadIdx.x;
  const int lane = tid & 31;
  const int wave = tid >> 5;
  const int wm = wave >> 1, wn = wave & 1;
  const int mBase = blockIdx.x * 128;
  const int nBase = blockIdx.y * 128;
  const int m_in = lane & 15;
  const int kb = (lane >> 4) ? 8 : 0;
  const int mb = (lane >> 4) * 8;

  unsigned int* lA32 = (unsigned int*)lA;
  unsigned int* lB32 = (unsigned int*)lB;

  if (tid < 128) lSum[tid] = 0.0f;

  v8f z = {0, 0, 0, 0, 0, 0, 0, 0};
  v8f acc[2][4];
#pragma unroll
  for (int i = 0; i < 2; ++i)
#pragma unroll
    for (int j = 0; j < 4; ++j) acc[i][j] = z;

  for (int kc = 0; kc < 256; kc += 32) {
    __syncthreads();
    if (kc + 32 < 256)
      __builtin_prefetch(&A[(size_t)mBase * 256 + kc + 32], 0, 0);
    // A tile 128x32: float2 global loads, packed b32 LDS stores (2048 pairs)
#pragma unroll
    for (int it = 0; it < 8; ++it) {
      int p = tid + it * 256;
      int r = p >> 4, k2 = p & 15;
      float2 f = *(const float2*)(A + (size_t)(mBase + r) * 256 + kc + 2 * k2);
      lA32[r * 16 + k2] = pack2bf(f.x, f.y);
    }
    // B tile 32x128 transposed into lB[n][k]; pack k-pairs (coalesced over n)
#pragma unroll
    for (int it = 0; it < 8; ++it) {
      int p = tid + it * 256;
      int k2 = p >> 7, n = p & 127;
      float f0 = W[(size_t)(kc + 2 * k2) * 256 + nBase + n];
      float f1 = W[(size_t)(kc + 2 * k2 + 1) * 256 + nBase + n];
      lB32[n * 16 + k2] = pack2bf(f0, f1);
    }
    __syncthreads();

    Frag16 fa[2], fb[4];
#pragma unroll
    for (int ti = 0; ti < 2; ++ti) {
      const unsigned short* p = lA + (wm * 32 + ti * 16 + m_in) * 32 + kb;
      fa[ti].h[0] = *(const u16x8*)(p);
      fa[ti].h[1] = *(const u16x8*)(p + 16);
    }
#pragma unroll
    for (int tj = 0; tj < 4; ++tj) {
      const unsigned short* p = lB + (wn * 64 + tj * 16 + m_in) * 32 + kb;
      fb[tj].h[0] = *(const u16x8*)(p);
      fb[tj].h[1] = *(const u16x8*)(p + 16);
    }
#pragma unroll
    for (int ti = 0; ti < 2; ++ti)
#pragma unroll
      for (int tj = 0; tj < 4; ++tj)
        acc[ti][tj] = __builtin_amdgcn_wmma_f32_16x16x32_bf16(
            false, fa[ti].v, false, fb[tj].v, (short)0, acc[ti][tj], false, false);
  }

#pragma unroll
  for (int ti = 0; ti < 2; ++ti)
#pragma unroll
    for (int tj = 0; tj < 4; ++tj) {
      int row0 = mBase + wm * 32 + ti * 16 + mb;
      int col  = nBase + wn * 64 + tj * 16 + m_in;
      float ss = 0.0f;
#pragma unroll
      for (int r = 0; r < 8; ++r) {
        float v = acc[ti][tj][r];
        O[(size_t)(row0 + r) * 256 + col] = f2bf(v);
        ss += v * v;
      }
      if (sumsq) atomicAdd(&lSum[wn * 64 + tj * 16 + m_in], ss);
    }
  if (sumsq) {
    __syncthreads();
    if (tid < 128) {
      int b = mBase >> 14;
      atomicAdd(&sumsq[b * 256 + nBase + tid], lSum[tid]);
    }
  }
}

// ---------------------------------------------------------------------------
// Per (b,h): Gram G[i][j] = sum_n k[i,n]*q[j,n] via WMMA (K=16384), then
// normalize by token-axis L2 norms, *rescale, softmax rows, and emit
// W2[b][h*32+j][n] = sum_i attn[i][j] * Wp[h*32+i][n]  (bf16).
// LDS layout [ch][n]: adjacent tokens adjacent -> token-paired b32 stores;
// thread halves split K vs Q staging.
// ---------------------------------------------------------------------------
__global__ __launch_bounds__(256) void k_gram_softmax_w2(
    const unsigned short* __restrict__ K16, const unsigned short* __restrict__ Q16,
    const float* __restrict__ ssq_k, const float* __restrict__ ssq_q,
    const float* __restrict__ rescale, const float* __restrict__ Wp,
    unsigned short* __restrict__ W2) {
  __shared__ __align__(16) unsigned short lK[32 * 256]; // [ch][n]
  __shared__ __align__(16) unsigned short lQ[32 * 256];
  __shared__ float sG[32 * 32];
  __shared__ float sNk[32], sNq[32];

  const int tid = threadIdx.x;
  const int lane = tid & 31;
  const int wave = tid >> 5;
  const int b = blockIdx.x >> 3;
  const int h = blockIdx.x & 7;
  const int m_in = lane & 15;
  const int kb = (lane >> 4) ? 8 : 0;
  const int mb = (lane >> 4) * 8;

  v8f z = {0, 0, 0, 0, 0, 0, 0, 0};
  v8f acc[2][2];
  acc[0][0] = z; acc[0][1] = z; acc[1][0] = z; acc[1][1] = z;
  for (int idx = tid; idx < 1024; idx += 256) sG[idx] = 0.0f;

  const size_t rowbase = (size_t)b * NTOK;
  const int half = tid >> 7;     // 0 -> stage K, 1 -> stage Q
  const int tp   = tid & 127;    // token-pair index within super-chunk

  for (int ns = 0; ns < NTOK; ns += 256) {
    __syncthreads();
    { // stage 256 tokens x 32 head-channels, transposed, token-paired b32
      const unsigned short* src =
          (half ? Q16 : K16) + ((rowbase + ns + 2 * tp) * 256 + h * 32);
      unsigned int* dst = (unsigned int*)(half ? lQ : lK);
      u16x8 r0[4], r1[4];
#pragma unroll
      for (int v4 = 0; v4 < 4; ++v4) {
        r0[v4] = *(const u16x8*)(src + v4 * 8);
        r1[v4] = *(const u16x8*)(src + 256 + v4 * 8);
      }
#pragma unroll
      for (int v4 = 0; v4 < 4; ++v4)
#pragma unroll
        for (int e = 0; e < 8; ++e) {
          int ch = v4 * 8 + e;
          dst[ch * 128 + tp] =
              (unsigned int)r0[v4][e] | ((unsigned int)r1[v4][e] << 16);
        }
    }
    __syncthreads();
    const int nsub = wave * 32;                 // each wave owns one K-chunk
    Frag16 fa[2], fb[2];
#pragma unroll
    for (int t2 = 0; t2 < 2; ++t2) {
      const unsigned short* pk = lK + (t2 * 16 + m_in) * 256 + nsub + kb;
      fa[t2].h[0] = *(const u16x8*)(pk);
      fa[t2].h[1] = *(const u16x8*)(pk + 16);
      const unsigned short* pq = lQ + (t2 * 16 + m_in) * 256 + nsub + kb;
      fb[t2].h[0] = *(const u16x8*)(pq);
      fb[t2].h[1] = *(const u16x8*)(pq + 16);
    }
#pragma unroll
    for (int ti = 0; ti < 2; ++ti)
#pragma unroll
      for (int tj = 0; tj < 2; ++tj)
        acc[ti][tj] = __builtin_amdgcn_wmma_f32_16x16x32_bf16(
            false, fa[ti].v, false, fb[tj].v, (short)0, acc[ti][tj], false, false);
  }
  __syncthreads();
#pragma unroll
  for (int ti = 0; ti < 2; ++ti)
#pragma unroll
    for (int tj = 0; tj < 2; ++tj)
#pragma unroll
      for (int r = 0; r < 8; ++r)
        atomicAdd(&sG[(ti * 16 + mb + r) * 32 + tj * 16 + m_in], acc[ti][tj][r]);
  __syncthreads();

  if (tid < 32) {
    sNk[tid] = fmaxf(sqrtf(ssq_k[b * 256 + h * 32 + tid]), 1e-12f);
    sNq[tid] = fmaxf(sqrtf(ssq_q[b * 256 + h * 32 + tid]), 1e-12f);
  }
  __syncthreads();
  if (tid < 32) {                               // softmax over j, row i = tid
    const int i = tid;
    const float sc = rescale[h];
    float rowv[32], mx = -3.4e38f;
#pragma unroll
    for (int j = 0; j < 32; ++j) {
      float a = sG[i * 32 + j] / (sNk[i] * sNq[j]) * sc;
      rowv[j] = a;
      mx = fmaxf(mx, a);
    }
    float s = 0.0f;
#pragma unroll
    for (int j = 0; j < 32; ++j) { rowv[j] = expf(rowv[j] - mx); s += rowv[j]; }
    float inv = 1.0f / s;
#pragma unroll
    for (int j = 0; j < 32; ++j) sG[i * 32 + j] = rowv[j] * inv;
  }
  __syncthreads();
  { // W2 build: thread = output column n
    const int n = tid;
    float wp[32];
#pragma unroll
    for (int i = 0; i < 32; ++i) wp[i] = Wp[(size_t)(h * 32 + i) * 256 + n];
    unsigned short* w2b = W2 + (size_t)b * 256 * 256;
    for (int j = 0; j < 32; ++j) {
      float s = 0.0f;
#pragma unroll
      for (int i = 0; i < 32; ++i) s += sG[i * 32 + j] * wp[i];
      w2b[(size_t)(h * 32 + j) * 256 + n] = f2bf(s);
    }
  }
}

// ---------------------------------------------------------------------------
// Depthwise 3x3 SAME convs (NHWC, C=256). Block = one (b,h,w), thread = channel.
// ---------------------------------------------------------------------------
__global__ __launch_bounds__(256) void k_dwconv1_gelu(
    const unsigned short* __restrict__ V16, const float* __restrict__ kern,
    float* __restrict__ out) {
  const int c = threadIdx.x;
  const int b = blockIdx.x >> 14;
  const int hw = blockIdx.x & 16383;
  const int hh = hw >> 7, ww = hw & 127;
  float acc = 0.0f;
#pragma unroll
  for (int dh = 0; dh < 3; ++dh)
#pragma unroll
    for (int dw = 0; dw < 3; ++dw) {
      int y = hh + dh - 1, x = ww + dw - 1;
      if (y >= 0 && y < 128 && x >= 0 && x < 128)
        acc += bf2f(V16[((size_t)(b << 14) + (y << 7) + x) * 256 + c]) *
               kern[(dh * 3 + dw) * 256 + c];
    }
  float g = 0.5f * acc * (1.0f + erff(acc * 0.70710678118654752f)); // exact gelu
  out[((size_t)(b << 14) + (hh << 7) + ww) * 256 + c] = g;
}

__global__ __launch_bounds__(256) void k_dwconv2(
    const float* __restrict__ in, const float* __restrict__ kern,
    float* __restrict__ out) {
  const int c = threadIdx.x;
  const int b = blockIdx.x >> 14;
  const int hw = blockIdx.x & 16383;
  const int hh = hw >> 7, ww = hw & 127;
  float acc = 0.0f;
#pragma unroll
  for (int dh = 0; dh < 3; ++dh)
#pragma unroll
    for (int dw = 0; dw < 3; ++dw) {
      int y = hh + dh - 1, x = ww + dw - 1;
      if (y >= 0 && y < 128 && x >= 0 && x < 128)
        acc += in[((size_t)(b << 14) + (y << 7) + x) * 256 + c] *
               kern[(dh * 3 + dw) * 256 + c];
    }
  out[((size_t)(b << 14) + (hh << 7) + ww) * 256 + c] = acc; // out_p
}

// ---------------------------------------------------------------------------
// Final GEMM: d_out[r][n] += v16[r][:] @ W2[b][:][n] + bp[n]  (out_p already in d_out)
// ---------------------------------------------------------------------------
__global__ __launch_bounds__(256) void k_final(
    const unsigned short* __restrict__ V16, const unsigned short* __restrict__ W2,
    const float* __restrict__ bp, float* __restrict__ out) {
  __shared__ __align__(16) unsigned short lA[128 * 32];
  __shared__ __align__(16) unsigned short lB[128 * 32];

  const int tid  = threadIdx.x;
  const int lane = tid & 31;
  const int wave = tid >> 5;
  const int wm = wave >> 1, wn = wave & 1;
  const int mBase = blockIdx.x * 128;
  const int nBase = blockIdx.y * 128;
  const int b = mBase >> 14;
  const int m_in = lane & 15;
  const int kb = (lane >> 4) ? 8 : 0;
  const int mb = (lane >> 4) * 8;
  const unsigned short* W2b = W2 + (size_t)b * 256 * 256;
  unsigned int* lB32 = (unsigned int*)lB;

  v8f z = {0, 0, 0, 0, 0, 0, 0, 0};
  v8f acc[2][4];
#pragma unroll
  for (int i = 0; i < 2; ++i)
#pragma unroll
    for (int j = 0; j < 4; ++j) acc[i][j] = z;

  for (int kc = 0; kc < 256; kc += 32) {
    __syncthreads();
    // A tile: straight 16B vector copies (bf16 already)
#pragma unroll
    for (int it = 0; it < 2; ++it) {
      int p = tid + it * 256;            // 512 vectors of 8 bf16
      int r = p >> 2, k8 = p & 3;
      *(u16x8*)(lA + r * 32 + k8 * 8) =
          *(const u16x8*)(V16 + (size_t)(mBase + r) * 256 + kc + k8 * 8);
    }
    // B tile transposed with packed k-pairs
#pragma unroll
    for (int it = 0; it < 8; ++it) {
      int p = tid + it * 256;
      int k2 = p >> 7, n = p & 127;
      unsigned int lo = W2b[(size_t)(kc + 2 * k2) * 256 + nBase + n];
      unsigned int hi = W2b[(size_t)(kc + 2 * k2 + 1) * 256 + nBase + n];
      lB32[n * 16 + k2] = lo | (hi << 16);
    }
    __syncthreads();

    Frag16 fa[2], fb[4];
#pragma unroll
    for (int ti = 0; ti < 2; ++ti) {
      const unsigned short* p = lA + (wm * 32 + ti * 16 + m_in) * 32 + kb;
      fa[ti].h[0] = *(const u16x8*)(p);
      fa[ti].h[1] = *(const u16x8*)(p + 16);
    }
#pragma unroll
    for (int tj = 0; tj < 4; ++tj) {
      const unsigned short* p = lB + (wn * 64 + tj * 16 + m_in) * 32 + kb;
      fb[tj].h[0] = *(const u16x8*)(p);
      fb[tj].h[1] = *(const u16x8*)(p + 16);
    }
#pragma unroll
    for (int ti = 0; ti < 2; ++ti)
#pragma unroll
      for (int tj = 0; tj < 4; ++tj)
        acc[ti][tj] = __builtin_amdgcn_wmma_f32_16x16x32_bf16(
            false, fa[ti].v, false, fb[tj].v, (short)0, acc[ti][tj], false, false);
  }

#pragma unroll
  for (int ti = 0; ti < 2; ++ti)
#pragma unroll
    for (int tj = 0; tj < 4; ++tj) {
      int row0 = mBase + wm * 32 + ti * 16 + mb;
      int col  = nBase + wn * 64 + tj * 16 + m_in;
      float bias = bp[col];
#pragma unroll
      for (int r = 0; r < 8; ++r) {
        size_t o = (size_t)(row0 + r) * 256 + col;
        out[o] = out[o] + acc[ti][tj][r] + bias;   // += out_c + bias (out_p present)
      }
    }
}

extern "C" void kernel_launch(void* const* d_in, const int* in_sizes, int n_in,
                              void* d_out, int out_size, void* d_ws, size_t ws_size,
                              hipStream_t stream) {
  const float* x_in    = (const float*)d_in[0];
  const float* y_in    = (const float*)d_in[1];
  const float* f_in    = (const float*)d_in[2];
  const float* Wq      = (const float*)d_in[3];
  const float* Wk      = (const float*)d_in[4];
  const float* Wv      = (const float*)d_in[5];
  const float* rescale = (const float*)d_in[6];
  const float* Wp      = (const float*)d_in[7];
  const float* bp      = (const float*)d_in[8];
  const float* pw1     = (const float*)d_in[9];
  const float* pw2     = (const float*)d_in[10];
  float* out = (float*)d_out;

  char* ws = (char*)d_ws;
  unsigned short* q16 = (unsigned short*)(ws);
  unsigned short* k16 = q16 + (size_t)ROWS * 256;
  unsigned short* v16 = k16 + (size_t)ROWS * 256;
  float* t1    = (float*)(v16 + (size_t)ROWS * 256);
  float* ssq_q = t1 + (size_t)ROWS * 256;
  float* ssq_k = ssq_q + BATCH * 256;
  unsigned short* W2 = (unsigned short*)(ssq_k + BATCH * 256);

  dim3 gGemm(ROWS / 128, 2, 1);

  k_zero<<<1, 256, 0, stream>>>(ssq_q, 2 * BATCH * 256);
  k_gemm_qkv<<<gGemm, 256, 0, stream>>>(x_in, Wq, q16, ssq_q);
  k_gemm_qkv<<<gGemm, 256, 0, stream>>>(y_in, Wk, k16, ssq_k);
  k_gemm_qkv<<<gGemm, 256, 0, stream>>>(f_in, Wv, v16, nullptr);
  k_gram_softmax_w2<<<BATCH * HEADS, 256, 0, stream>>>(k16, q16, ssq_k, ssq_q,
                                                       rescale, Wp, W2);
  k_dwconv1_gelu<<<BATCH * NTOK, 256, 0, stream>>>(v16, pw1, t1);
  k_dwconv2<<<BATCH * NTOK, 256, 0, stream>>>(t1, pw2, out);   // d_out = out_p
  k_final<<<gGemm, 256, 0, stream>>>(v16, W2, bp, out);        // d_out += out_c + bp
}
